// SimReg_28226525070029
// MI455X (gfx1250) — matrix operand order repaired and checked by reading, not compiled, not run on hardware
//
#include <hip/hip_runtime.h>

// LNCC(win=7) fused single-pass kernel for MI455X (gfx1250, wave32).
// Bandwidth-bound: 78.6 MB in -> 1 float out, HBM floor ~3.4us @ 23.3 TB/s.
// One pass: halo'd 22^3 tile in LDS; the three separable 7-tap box sums are
// banded-ones matmuls on the WMMA units (v_wmma_f32_16x16x32_f16, K=32 covers
// the 22-wide halo'd span in ONE wmma per 16x16 block). All LDS layouts are
// padded (stride 24) so operand gathers are unconditional ds_load_b128 and
// stores are unconditional packed stores -> no EXEC churn. Field loop is
// unrolled (compile-time specialization); the cc math + reduction is folded
// into the last field's Z-pass. Deterministic two-stage final reduce.

typedef __attribute__((ext_vector_type(16))) _Float16 v16h;
typedef __attribute__((ext_vector_type(8)))  _Float16 v8h;
typedef __attribute__((ext_vector_type(2)))  _Float16 h2;
typedef __attribute__((ext_vector_type(8)))  float    v8f;
typedef __attribute__((ext_vector_type(4)))  float    v4f;

#define DZ 160
#define DY 192
#define DX 160
#define TILE 16
#define TIN 22                   // TILE + 2*3 halo
#define XS  24                   // padded inner stride (halfs)
#define NTX 10
#define NTY 12
#define NTZ 10
#define NBLOCKS 2400             // 2 batches * 10*12*10 tiles
#define WSZ  343.0f
#define NVOX 9830400.0f

#define SI_HALFS (TIN*TIN*XS)    // 11616
#define TA_HALFS (TIN*TILE*XS)   // 8448
#define TB_HALFS (TILE*TILE*XS)  // 6144

// Manual LDS pool: guarantees f16 buffers are adjacent (padded/overflow reads
// always hit finite f16 data, never f32 bit patterns), guard before res.
#define OFF_SI    0
#define OFF_SJ    (OFF_SI + SI_HALFS*2)     // 23232
#define OFF_TA    (OFF_SJ + SI_HALFS*2)     // 46464
#define OFF_TB    (OFF_TA + TA_HALFS*2)     // 63360
#define OFF_GUARD (OFF_TB + TB_HALFS*2)     // 75648
#define OFF_RES   (OFF_GUARD + 32)          // 75680 (16B aligned)
#define OFF_RED   (OFF_RES + 4*4096*4)      // 141216
#define SMEM_BYTES (OFF_RED + 256*4)        // 142240 (~139 KB)

// A-operand (16x32 f16) element e -> K, per CDNA5 ISA 7.12.2
__device__ __forceinline__ int a_kidx(int e, int hi) {
    return (e & 7) + ((e >> 3) << 4) + (hi << 3);
}
// Contiguous K order (Y/Z-pass operands): lanes0-15 K=0..15, lanes16-31 K=16..31
__device__ __forceinline__ int b_kidx(int e, int hi) {
    return e + (hi << 4);
}
__device__ __forceinline__ v16h cat16(v8h a, v8h b) {
    return __builtin_shufflevector(a, b, 0,1,2,3,4,5,6,7,8,9,10,11,12,13,14,15);
}

__global__ __launch_bounds__(256) void lncc_tile_kernel(
        const float* __restrict__ gI, const float* __restrict__ gJ,
        float* __restrict__ partials) {
    __shared__ __align__(64) unsigned char smem[SMEM_BYTES];
    _Float16* sI  = (_Float16*)(smem + OFF_SI);
    _Float16* sJ  = (_Float16*)(smem + OFF_SJ);
    _Float16* tA  = (_Float16*)(smem + OFF_TA);  // x-summed  [z][x][y pad24]
    _Float16* tB  = (_Float16*)(smem + OFF_TB);  // xy-summed [y'][x][z pad24]
    float*    res0 = (float*)(smem + OFF_RES);             // I_sum  [x][y][z]
    float*    res1 = res0 + 4096;                          // J_sum
    float*    res2 = res1 + 4096;                          // cross
    float*    res3 = res2 + 4096;                          // I_var
    float*    red  = (float*)(smem + OFF_RED);

    const int tid  = threadIdx.x;
    const int lane = tid & 31;   // wave32
    const int lo   = lane & 15;  // M (A) / N (B,D)
    const int hi   = lane >> 4;
    const int wv   = tid >> 5;   // wave id 0..7

    const int blk = blockIdx.x;
    const int tx = blk % NTX;
    const int ty = (blk / NTX) % NTY;
    const int tz = (blk / (NTX * NTY)) % NTZ;
    const int bb_ = blk / (NTX * NTY * NTZ);
    const int x0 = tx * TILE - 3, y0 = ty * TILE - 3, z0 = tz * TILE - 3;
    const size_t base = (size_t)bb_ * DZ * DY * DX;

    // ---- load halo'd tiles (zero-padded borders + x-pad), f32 -> f16 ----
    for (int i = tid; i < SI_HALFS; i += 256) {
        int x = i % XS, y = (i / XS) % TIN, z = i / (XS * TIN);
        int gx = x0 + x, gy = y0 + y, gz = z0 + z;
        float vi = 0.f, vj = 0.f;
        if (x < TIN && (unsigned)gx < DX && (unsigned)gy < DY && (unsigned)gz < DZ) {
            size_t g = base + ((size_t)gz * DY + gy) * DX + gx;
            vi = gI[g]; vj = gJ[g];
        }
        sI[i] = (_Float16)vi;
        sJ[i] = (_Float16)vj;
    }
    // zero tA, tB, guard (pads must be exactly 0.0h: WMMA propagates 0*NaN)
    {
        _Float16* zp = tA;
        for (int i = tid; i < TA_HALFS + TB_HALFS + 16; i += 256)
            zp[i] = (_Float16)0.0f;
    }
    __syncthreads();

    // Banded-ones operands (registers, built once)
    v16h bandA;  // A[m,k] = 1 iff m <= k <= m+6   (Y/Z passes)
    v16h bandB;  // B[k,n] = 1 iff n <= k <= n+6   (X pass)
    #pragma unroll
    for (int e = 0; e < 16; ++e) {
        int ka = b_kidx(e, hi);
        int kb = a_kidx(e, hi);
        bandA[e] = (ka >= lo && ka <= lo + 6) ? (_Float16)1.0f : (_Float16)0.0f;
        bandB[e] = (kb >= lo && kb <= lo + 6) ? (_Float16)1.0f : (_Float16)0.0f;
    }

    float acc = 0.f;

    // field order: 0=I, 1=J, 2=I*J, 3=I*I, 4=J*J
    #pragma unroll
    for (int f = 0; f < 5; ++f) {
        // ---- X pass: D = DataRows x BandB. 22 z * 2 overlapping y-blocks ----
        for (int t = wv; t < 2 * TIN; t += 8) {
            const int z = t >> 1, yb = (t & 1) * 6;       // rows 0..15 / 6..21
            const int row = (z * TIN + (yb + lo)) * XS;
            v8h a0, a1;
            if (f == 0) {
                a0 = *(const v8h*)&sI[row + hi * 8];
                a1 = *(const v8h*)&sI[row + 16 + hi * 8];
            } else if (f == 1) {
                a0 = *(const v8h*)&sJ[row + hi * 8];
                a1 = *(const v8h*)&sJ[row + 16 + hi * 8];
            } else if (f == 2) {
                v8h i0 = *(const v8h*)&sI[row + hi * 8];
                v8h i1 = *(const v8h*)&sI[row + 16 + hi * 8];
                v8h j0 = *(const v8h*)&sJ[row + hi * 8];
                v8h j1 = *(const v8h*)&sJ[row + 16 + hi * 8];
                a0 = i0 * j0; a1 = i1 * j1;               // v_pk_mul_f16
            } else if (f == 3) {
                v8h i0 = *(const v8h*)&sI[row + hi * 8];
                v8h i1 = *(const v8h*)&sI[row + 16 + hi * 8];
                a0 = i0 * i0; a1 = i1 * i1;
            } else {
                v8h j0 = *(const v8h*)&sJ[row + hi * 8];
                v8h j1 = *(const v8h*)&sJ[row + 16 + hi * 8];
                a0 = j0 * j0; a1 = j1 * j1;
            }
            v8f c = {};
            c = __builtin_amdgcn_wmma_f32_16x16x32_f16(
                    false, cat16(a0, a1), false, bandB, (short)0, c, false, false);
            // tA[z][x=lo][y] ; rows m -> y = yb + m, packed f16 pairs (b32)
            const int sb = (z * TILE + lo) * XS + yb + 8 * hi;
            #pragma unroll
            for (int v = 0; v < 8; v += 2) {
                h2 p = { (_Float16)c[v], (_Float16)c[v + 1] };
                *(h2*)&tA[sb + v] = p;
            }
        }
        __syncthreads();

        // ---- Y pass: D = BandA x tA_cols. 22 z-slices ----
        for (int z = wv; z < TIN; z += 8) {
            const int rb = (z * TILE + lo) * XS + hi * 16;
            v16h b = cat16(*(const v8h*)&tA[rb], *(const v8h*)&tA[rb + 8]);
            v8f c = {};
            c = __builtin_amdgcn_wmma_f32_16x16x32_f16(
                    false, bandA, false, b, (short)0, c, false, false);
            // tB[y'=m][x=lo][z]
            #pragma unroll
            for (int v = 0; v < 8; ++v) {
                int m = v + 8 * hi;
                tB[(m * TILE + lo) * XS + z] = (_Float16)c[v];
            }
        }
        __syncthreads();

        // ---- Z pass: D = BandA x tB_cols. 16 y-slices ----
        for (int y = wv; y < TILE; y += 8) {
            const int rb = (y * TILE + lo) * XS + hi * 16;
            v16h b = cat16(*(const v8h*)&tB[rb], *(const v8h*)&tB[rb + 8]);
            v8f c = {};
            c = __builtin_amdgcn_wmma_f32_16x16x32_f16(
                    false, bandA, false, b, (short)0, c, false, false);
            const int rbase = (lo * TILE + y) * TILE + 8 * hi;  // res [x][y][z']
            v4f c0 = { c[0], c[1], c[2], c[3] };
            v4f c1 = { c[4], c[5], c[6], c[7] };
            if (f == 0) {
                *(v4f*)&res0[rbase] = c0; *(v4f*)&res0[rbase + 4] = c1;
            } else if (f == 1) {
                *(v4f*)&res1[rbase] = c0; *(v4f*)&res1[rbase + 4] = c1;
            } else if (f == 2) {
                // cross = IJ_sum - I_sum*J_sum/343
                v4f Is0 = *(v4f*)&res0[rbase], Is1 = *(v4f*)&res0[rbase + 4];
                v4f Js0 = *(v4f*)&res1[rbase], Js1 = *(v4f*)&res1[rbase + 4];
                *(v4f*)&res2[rbase]     = c0 - Is0 * Js0 * (1.0f / WSZ);
                *(v4f*)&res2[rbase + 4] = c1 - Is1 * Js1 * (1.0f / WSZ);
            } else if (f == 3) {
                // I_var = I2_sum - I_sum^2/343
                v4f Is0 = *(v4f*)&res0[rbase], Is1 = *(v4f*)&res0[rbase + 4];
                *(v4f*)&res3[rbase]     = c0 - Is0 * Is0 * (1.0f / WSZ);
                *(v4f*)&res3[rbase + 4] = c1 - Is1 * Is1 * (1.0f / WSZ);
            } else {
                // J_var = J2_sum - J_sum^2/343; cc = (cross^2+0.1)/(Ivar*Jvar+0.1)
                v4f Js0 = *(v4f*)&res1[rbase], Js1 = *(v4f*)&res1[rbase + 4];
                v4f Jv0 = c0 - Js0 * Js0 * (1.0f / WSZ);
                v4f Jv1 = c1 - Js1 * Js1 * (1.0f / WSZ);
                v4f cr0 = *(v4f*)&res2[rbase], cr1 = *(v4f*)&res2[rbase + 4];
                v4f Iv0 = *(v4f*)&res3[rbase], Iv1 = *(v4f*)&res3[rbase + 4];
                v4f cc0 = (cr0 * cr0 + 0.1f) / (Iv0 * Jv0 + 0.1f);
                v4f cc1 = (cr1 * cr1 + 0.1f) / (Iv1 * Jv1 + 0.1f);
                acc += cc0[0] + cc0[1] + cc0[2] + cc0[3]
                     + cc1[0] + cc1[1] + cc1[2] + cc1[3];
            }
        }
        __syncthreads();
    }

    // ---- block reduction ----
    red[tid] = acc;
    __syncthreads();
    for (int off = 128; off > 0; off >>= 1) {
        if (tid < off) red[tid] += red[tid + off];
        __syncthreads();
    }
    if (tid == 0) partials[blockIdx.x] = red[0];
}

__global__ __launch_bounds__(256) void lncc_reduce_kernel(
        const float* __restrict__ partials, float* __restrict__ out) {
    __shared__ float s[256];
    float acc = 0.f;
    for (int i = threadIdx.x; i < NBLOCKS; i += 256) acc += partials[i];
    s[threadIdx.x] = acc;
    __syncthreads();
    for (int off = 128; off > 0; off >>= 1) {
        if (threadIdx.x < off) s[threadIdx.x] += s[threadIdx.x + off];
        __syncthreads();
    }
    if (threadIdx.x == 0) out[0] = s[0] / NVOX;
}

extern "C" void kernel_launch(void* const* d_in, const int* in_sizes, int n_in,
                              void* d_out, int out_size, void* d_ws, size_t ws_size,
                              hipStream_t stream) {
    (void)in_sizes; (void)n_in; (void)out_size; (void)ws_size;
    const float* I = (const float*)d_in[0];   // x2y
    const float* J = (const float*)d_in[1];   // fixed
    float* partials = (float*)d_ws;           // NBLOCKS floats
    float* out = (float*)d_out;

    lncc_tile_kernel<<<NBLOCKS, 256, 0, stream>>>(I, J, partials);
    lncc_reduce_kernel<<<1, 256, 0, stream>>>(partials, out);
}
